// SpikingNeuralNetwork_85899346051
// MI455X (gfx1250) — matrix operand order repaired and checked by reading, not compiled, not run on hardware
//
#include <hip/hip_runtime.h>

// CDNA5 / gfx1250 — wave32, WMMA f32 = f16 x f16 (16x16x32), f32 accumulate.
typedef __attribute__((ext_vector_type(16))) _Float16 v16h;
typedef __attribute__((ext_vector_type(8)))  _Float16 v8h;
typedef __attribute__((ext_vector_type(8)))  float    v8f;

#define SNN_B 64
#define SNN_T 256

// ---------------------------------------------------------------------------
// Tiled transpose + f32->f16 downconvert:  in [Z][R][C] f32 -> out [Z][C][R] f16
// ---------------------------------------------------------------------------
__global__ __launch_bounds__(256)
void transpose_to_f16(const float* __restrict__ in, _Float16* __restrict__ out,
                      int R, int C, size_t in_zs, size_t out_zs)
{
    __shared__ float tile[32][33];
    const int z  = blockIdx.z;
    const int r0 = blockIdx.y * 32;
    const int c0 = blockIdx.x * 32;
    const int lr = threadIdx.x >> 5;   // 0..7
    const int lc = threadIdx.x & 31;

    const float* src = in + (size_t)z * in_zs;
    #pragma unroll
    for (int s = 0; s < 4; ++s) {
        const int r = lr + s * 8;
        tile[r][lc] = src[(size_t)(r0 + r) * C + (c0 + lc)];   // coalesced in C
    }
    __syncthreads();
    _Float16* dst = out + (size_t)z * out_zs;
    #pragma unroll
    for (int s = 0; s < 4; ++s) {
        const int c = lr + s * 8;
        dst[(size_t)(c0 + c) * R + (r0 + lc)] = (_Float16)tile[lc][c]; // coalesced in R
    }
}

// ---------------------------------------------------------------------------
// Fused layer: u = W^T [x;1], then LIF scan over T.
// Inputs are K-contiguous f16: xt [B][T][N_IN], Wt [N_OUT][N_IN].
// Block = 256 thr (8 waves): one batch b, 32 output rows, all 256 timesteps.
// Wave w: M-subtile (w&1), t-columns [(w>>1)*64, +64).
// ---------------------------------------------------------------------------
template<int N_IN, int N_OUT, bool LAST>
__global__ __launch_bounds__(256)
void snn_layer(const _Float16* __restrict__ xt,   // [B, T, N_IN] f16
               const _Float16* __restrict__ Wt,   // [N_OUT, N_IN] f16
               const float* __restrict__ Wb,      // original W: bias row N_IN
               const float* __restrict__ beta_p,
               const float* __restrict__ thr_p,
               _Float16* __restrict__ h_out,      // [B, T, N_OUT] f16 (hidden)
               float* __restrict__ spk_out,       // [B, N_OUT, T] f32 (last)
               float* __restrict__ mem_out)       // [B, N_OUT, T] f32 (last)
{
    __shared__ float    u_tile[32][SNN_T + 1];    // padded: conflict-free scan
    __shared__ _Float16 s_tile[32][SNN_T + 2];

    const int tid  = threadIdx.x;
    const int lane = tid & 31;
    const int w    = tid >> 5;
    const int half = lane >> 4;
    const int l15  = lane & 15;

    const int b       = blockIdx.y;
    const int o_block = blockIdx.x * 32;
    const int m_local = (w & 1) * 16;
    const int t_base  = (w >> 1) * 64;

    // ---- bias row (k == N_IN of original f32 W) initializes accumulators --
    v8f acc[4];
    {
        const size_t bias_base =
            (size_t)N_IN * N_OUT + o_block + m_local + 8 * half;
        float bias[8];
        #pragma unroll
        for (int r = 0; r < 8; ++r) bias[r] = Wb[bias_base + r];
        #pragma unroll
        for (int j = 0; j < 4; ++j) {
            #pragma unroll
            for (int r = 0; r < 8; ++r) acc[j][r] = bias[r];
        }
    }

    // ---- K loop: all operands are contiguous 16B vector loads --------------
    const int o_col = o_block + m_local + l15;                // A: M = l15
    const _Float16* arow = Wt + (size_t)o_col * N_IN;
    const _Float16* brow = xt + ((size_t)b * SNN_T + t_base + l15) * N_IN
                              + half * 16;                    // B: N = l15

    for (int k0 = 0; k0 < N_IN; k0 += 32) {
        // A fragment: lane holds 16 contiguous K in [half*16,+16)
        const v8h alo = *(const v8h*)(arow + k0 + half * 16);
        const v8h ahi = *(const v8h*)(arow + k0 + half * 16 + 8);
        const v16h a = __builtin_shufflevector(alo, ahi,
                          0,1,2,3,4,5,6,7,8,9,10,11,12,13,14,15);
        #pragma unroll
        for (int j = 0; j < 4; ++j) {
            // B fragment: lane holds 16 contiguous K at its own t row
            const _Float16* bp = brow + (size_t)(16 * j) * N_IN + k0;
            const v8h blo = *(const v8h*)(bp);
            const v8h bhi = *(const v8h*)(bp + 8);
            const v16h bf = __builtin_shufflevector(blo, bhi,
                              0,1,2,3,4,5,6,7,8,9,10,11,12,13,14,15);
            acc[j] = __builtin_amdgcn_wmma_f32_16x16x32_f16(
                         false, a, false, bf, (short)0, acc[j], false, false);
        }
    }

    // ---- spill D fragments (lane l, vgpr r -> M=8*half+r, N=l15) ----------
    #pragma unroll
    for (int j = 0; j < 4; ++j) {
        #pragma unroll
        for (int r = 0; r < 8; ++r)
            u_tile[m_local + 8 * half + r][t_base + 16 * j + l15] = acc[j][r];
    }
    __syncthreads();

    // ---- LIF scan: 32 rows, one lane per row, sequential in T --------------
    if (tid < 32) {
        const float beta = beta_p[0];
        const float thr  = thr_p[0];
        const float sb   = 1.0f / (1.0f + expf(-beta));   // sigmoid(beta)
        float mem = 0.0f;
        const int r = tid;
        for (int t = 0; t < SNN_T; ++t) {
            const float u      = u_tile[r][t];
            const float mem_in = mem;                  // state entering step t
            const float mem_u  = sb * mem + u;         // StateUpdate
            const float spike  = (mem_u >= thr) ? 1.0f : 0.0f;
            mem = mem_u - spike * thr;                 // StatePostupdate
            s_tile[r][t] = (_Float16)spike;
            u_tile[r][t] = mem_in;                     // mems (used when LAST)
        }
    }
    __syncthreads();

    // ---- writeout ----------------------------------------------------------
    if (LAST) {
        // reference layout [B, N_OUT, T]: iter j -> row j, t = tid (coalesced)
        #pragma unroll 4
        for (int j = 0; j < 32; ++j) {
            const size_t gi = ((size_t)b * N_OUT + o_block + j) * SNN_T + tid;
            spk_out[gi] = (float)s_tile[j][tid];
            mem_out[gi] = u_tile[j][tid];
        }
    } else {
        // transposed hidden layout [B, T, N_OUT]: lane = o (coalesced halves)
        const int o = tid & 31;
        const int tw = tid >> 5;                       // 0..7
        #pragma unroll 4
        for (int j = 0; j < 32; ++j) {
            const int t = j * 8 + tw;
            h_out[((size_t)b * SNN_T + t) * N_OUT + o_block + o] = s_tile[o][t];
        }
    }
}

extern "C" void kernel_launch(void* const* d_in, const int* in_sizes, int n_in,
                              void* d_out, int out_size, void* d_ws, size_t ws_size,
                              hipStream_t stream)
{
    (void)in_sizes; (void)n_in; (void)out_size; (void)ws_size;

    const float* x    = (const float*)d_in[0];  // [64, 512, 256]
    const float* W0   = (const float*)d_in[1];  // [513, 1024]
    const float* W1   = (const float*)d_in[2];  // [1025, 1024]
    const float* W2   = (const float*)d_in[3];  // [1025, 256]
    const float* beta = (const float*)d_in[4];
    const float* thr  = (const float*)d_in[5];

    _Float16* xt  = (_Float16*)d_ws;                              // [64,256,512]
    _Float16* h0t = xt  + (size_t)SNN_B * SNN_T * 512;            // [64,256,1024]
    _Float16* h1t = h0t + (size_t)SNN_B * SNN_T * 1024;           // [64,256,1024]
    _Float16* wt0 = h1t + (size_t)SNN_B * SNN_T * 1024;           // [1024,512]
    _Float16* wt1 = wt0 + (size_t)1024 * 512;                     // [1024,1024]
    _Float16* wt2 = wt1 + (size_t)1024 * 1024;                    // [256,1024]

    float* spk = (float*)d_out;
    float* mem = spk + (size_t)SNN_B * 256 * SNN_T;

    const dim3 blk(256);

    // Pre-transpose to K-contiguous f16 (bias rows of W stay in f32 source).
    transpose_to_f16<<<dim3(1024/32, 512/32, 1),  blk, 0, stream>>>(W0, wt0,  512, 1024, 0, 0);
    transpose_to_f16<<<dim3(1024/32, 1024/32, 1), blk, 0, stream>>>(W1, wt1, 1024, 1024, 0, 0);
    transpose_to_f16<<<dim3( 256/32, 1024/32, 1), blk, 0, stream>>>(W2, wt2, 1024,  256, 0, 0);
    transpose_to_f16<<<dim3( 256/32, 512/32, SNN_B), blk, 0, stream>>>(
        x, xt, 512, 256, (size_t)512 * SNN_T, (size_t)SNN_T * 512);

    snn_layer< 512, 1024, false><<<dim3(1024/32, SNN_B), blk, 0, stream>>>(
        xt,  wt0, W0, beta, thr, h0t, nullptr, nullptr);
    snn_layer<1024, 1024, false><<<dim3(1024/32, SNN_B), blk, 0, stream>>>(
        h0t, wt1, W1, beta, thr, h1t, nullptr, nullptr);
    snn_layer<1024,  256, true ><<<dim3( 256/32, SNN_B), blk, 0, stream>>>(
        h1t, wt2, W2, beta, thr, nullptr, spk, mem);
}
